// VGGTLayer_21569325760884
// MI455X (gfx1250) — matrix operand order repaired and checked by reading, not compiled, not run on hardware
//
#include <hip/hip_runtime.h>
#include <math.h>

// Problem constants (match reference)
#define B_      4
#define N_      2048
#define D_      256
#define H_      8
#define DH_     32
#define MLPH_   1024
#define E_      262144
#define BNROWS  (B_ * N_)      // 8192
#define EPS_    1e-5f

// Padded LDS row length (in ushorts): 80 bytes -> keeps 16B alignment for
// b128 fragment loads while avoiding power-of-two bank strides.
#define LPAD 40

typedef __attribute__((ext_vector_type(16))) __bf16        v16bf;
typedef __attribute__((ext_vector_type(8)))  float         v8f;
typedef __attribute__((ext_vector_type(4)))  float         fv4;
typedef __attribute__((ext_vector_type(4)))  unsigned int  uiv4;
typedef __attribute__((ext_vector_type(2)))  unsigned int  uiv2;

union FragBF {
  unsigned short u[16];
  unsigned int   u32[8];
  uiv4           q[2];
  v16bf          v;
};

// f32 -> bf16 round-to-nearest-even
__device__ __forceinline__ unsigned short f2bf(float f) {
  unsigned int u = __float_as_uint(f);
  unsigned int r = u + 0x7FFFu + ((u >> 16) & 1u);
  return (unsigned short)(r >> 16);
}
__device__ __forceinline__ unsigned int pack2bf(float a, float b) {
  return (unsigned int)f2bf(a) | ((unsigned int)f2bf(b) << 16);
}

// Fragment layout notes (ISA 7.12.2, 16-bit data):
//  A (16x32, MxK): lane m = lane&15; kbase = (lane&16)?8:0.
//    u[0..7]  = K = kbase+0 .. kbase+7        (contiguous)
//    u[8..15] = K = 16+kbase+0 .. 16+kbase+7  (contiguous)
//  B (32x16, KxN): lane n = lane&15; same K order as A (kbase).
//    u[0..7] = K = kbase+0..7, u[8..15] = K = 16+kbase+0..7
//  C/D (16x16 f32): VGPR r -> row r (lanes 0-15) / r+8 (lanes 16-31), col lane&15.

// ---------------------------------------------------------------------------
// Generic GEMM: out[M,Nn] = act(A[M,K] @ W[K,Nn] + bias), bf16 WMMA, f32 accum
// Block: 256 thr = 8 waves. Tile: BM=128 (wave w owns rows w*16..), BN=64, BK=32
// act: 0 = none, 1 = exact GELU
// ---------------------------------------------------------------------------
__global__ __launch_bounds__(256)
void gemm_bias_act(const float* __restrict__ A, const float* __restrict__ W,
                   const float* __restrict__ bias, float* __restrict__ out,
                   int M, int K, int Nn, int act) {
  __shared__ __align__(16) unsigned short sA[128 * LPAD];   // [m][k] bf16, K-major
  __shared__ __align__(16) unsigned short sBt[64 * LPAD];   // [n][k] bf16, transposed
  const int tid  = threadIdx.x;
  const int lane = tid & 31;
  const int w    = tid >> 5;
  const int n0   = blockIdx.x * 64;
  const int m0   = blockIdx.y * 128;

  v8f acc[4];
  #pragma unroll
  for (int t = 0; t < 4; ++t)
    #pragma unroll
    for (int r = 0; r < 8; ++r) acc[t][r] = 0.f;

  const int kb   = (lane & 16) ? 8 : 0;    // A fragment K base
  const int koff = (lane & 16) ? 16 : 0;   // unused: B fragments use kb
  const int arow = w * 16 + (lane & 15);

  for (int k0 = 0; k0 < K; k0 += 32) {
    // Stage A tile 128x32 via float4 loads (4 per thread), packed bf16 b64 stores
    #pragma unroll
    for (int i = 0; i < 4; ++i) {
      int lin4 = i * 256 + tid;            // 1024 float4 total
      int m = lin4 >> 3;
      int k = (lin4 & 7) * 4;
      fv4 f = *(const fv4*)(A + (size_t)(m0 + m) * K + k0 + k);
      uiv2 pk = { pack2bf(f.x, f.y), pack2bf(f.z, f.w) };
      *(uiv2*)&sA[m * LPAD + k] = pk;
    }
    // Stage B tile 32x64 transposed: coalesced global reads, scattered u16 stores
    #pragma unroll
    for (int i = 0; i < 8; ++i) {
      int lin = i * 256 + tid;             // 2048 elements
      int kk = lin >> 6, n = lin & 63;
      sBt[n * LPAD + kk] = f2bf(W[(size_t)(k0 + kk) * Nn + n0 + n]);
    }
    __syncthreads();

    // A fragment: two b128 LDS loads
    FragBF fa;
    fa.q[0] = *(const uiv4*)&sA[arow * LPAD + kb];
    fa.q[1] = *(const uiv4*)&sA[arow * LPAD + 16 + kb];

    #pragma unroll
    for (int t = 0; t < 4; ++t) {
      FragBF fb;
      int n = t * 16 + (lane & 15);
      fb.q[0] = *(const uiv4*)&sBt[n * LPAD + kb];
      fb.q[1] = *(const uiv4*)&sBt[n * LPAD + 16 + kb];
      acc[t] = __builtin_amdgcn_wmma_f32_16x16x32_bf16(
          false, fa.v, false, fb.v, (short)0, acc[t], false, false);
    }
    __syncthreads();
  }

  // Epilogue (C layout), fused bias + optional exact GELU
  #pragma unroll
  for (int t = 0; t < 4; ++t) {
    int n = n0 + t * 16 + (lane & 15);
    float bv = bias ? bias[n] : 0.f;
    #pragma unroll
    for (int r = 0; r < 8; ++r) {
      int m = m0 + w * 16 + r + ((lane & 16) ? 8 : 0);
      float val = acc[t][r] + bv;
      if (act == 1) val = 0.5f * val * (1.f + erff(val * 0.70710678118654752f));
      out[(size_t)m * Nn + n] = val;
    }
  }
}

// ---------------------------------------------------------------------------
// Flash attention: one wave per (b, h, q-tile of 16 rows). DH=32.
// K staged [key][d] (K-major for S=Q@K^T), V staged transposed [d][key]
// (K-major for P@V) -> all fragments are 2x ds_load_b128.
// ---------------------------------------------------------------------------
__global__ __launch_bounds__(256)
void attn_kernel(const float* __restrict__ Q, const float* __restrict__ K,
                 const float* __restrict__ V, float* __restrict__ O) {
  __shared__ __align__(16) unsigned short sK [8][32 * LPAD];  // [key][d]
  __shared__ __align__(16) unsigned short sVt[8][32 * LPAD];  // [d][key]
  __shared__ __align__(16) unsigned short sP [8][16 * LPAD];  // [qrow][key]
  const int tid  = threadIdx.x;
  const int lane = tid & 31;
  const int w    = tid >> 5;
  const int gw   = blockIdx.x * 8 + w;
  const int qt   = gw & (N_ / 16 - 1);   // 128 q tiles
  const int bh   = gw >> 7;
  const int b    = bh >> 3;
  const int h    = bh & (H_ - 1);
  const float scale = 0.17677669529663687f;  // 1/sqrt(DH)

  const size_t base = ((size_t)b * N_) * D_ + (size_t)h * DH_;
  const int kb   = (lane & 16) ? 8 : 0;
  const int koff = (lane & 16) ? 16 : 0;
  const int col  = lane & 15;

  // Persistent Q fragment from 4 float4 global loads (scale folded in)
  FragBF fq;
  {
    const float* qp = Q + base + (size_t)(qt * 16 + col) * D_;
    fv4 a0 = *(const fv4*)(qp + kb);
    fv4 a1 = *(const fv4*)(qp + kb + 4);
    fv4 a2 = *(const fv4*)(qp + 16 + kb);
    fv4 a3 = *(const fv4*)(qp + 16 + kb + 4);
    fq.u32[0] = pack2bf(scale * a0.x, scale * a0.y);
    fq.u32[1] = pack2bf(scale * a0.z, scale * a0.w);
    fq.u32[2] = pack2bf(scale * a1.x, scale * a1.y);
    fq.u32[3] = pack2bf(scale * a1.z, scale * a1.w);
    fq.u32[4] = pack2bf(scale * a2.x, scale * a2.y);
    fq.u32[5] = pack2bf(scale * a2.z, scale * a2.w);
    fq.u32[6] = pack2bf(scale * a3.x, scale * a3.y);
    fq.u32[7] = pack2bf(scale * a3.z, scale * a3.w);
  }

  v8f o0, o1;
  float m_st[8], l_st[8];
  #pragma unroll
  for (int r = 0; r < 8; ++r) { o0[r] = 0.f; o1[r] = 0.f; m_st[r] = -3.0e38f; l_st[r] = 0.f; }

  for (int kt = 0; kt < N_ / 32; ++kt) {
    // Stage K/V tiles: 256 float4 each, 8 per lane
    #pragma unroll
    for (int i = 0; i < 8; ++i) {
      int lin4 = i * 32 + lane;
      int key = lin4 >> 3;
      int d   = (lin4 & 7) * 4;
      size_t g = base + (size_t)(kt * 32 + key) * D_ + d;
      fv4 f = *(const fv4*)(K + g);
      uiv2 pk = { pack2bf(f.x, f.y), pack2bf(f.z, f.w) };
      *(uiv2*)&sK[w][key * LPAD + d] = pk;
      fv4 gv = *(const fv4*)(V + g);
      sVt[w][(d + 0) * LPAD + key] = f2bf(gv.x);
      sVt[w][(d + 1) * LPAD + key] = f2bf(gv.y);
      sVt[w][(d + 2) * LPAD + key] = f2bf(gv.z);
      sVt[w][(d + 3) * LPAD + key] = f2bf(gv.w);
    }
    asm volatile("s_wait_dscnt 0" ::: "memory");

    // S = Q @ K^T (B column n = key, K-dim = d)
    v8f s0, s1;
    {
      FragBF fb0, fb1;
      fb0.q[0] = *(const uiv4*)&sK[w][col * LPAD + kb];
      fb0.q[1] = *(const uiv4*)&sK[w][col * LPAD + 16 + kb];
      fb1.q[0] = *(const uiv4*)&sK[w][(16 + col) * LPAD + kb];
      fb1.q[1] = *(const uiv4*)&sK[w][(16 + col) * LPAD + 16 + kb];
      v8f z;
      #pragma unroll
      for (int r = 0; r < 8; ++r) z[r] = 0.f;
      s0 = __builtin_amdgcn_wmma_f32_16x16x32_bf16(false, fq.v, false, fb0.v, (short)0, z, false, false);
      s1 = __builtin_amdgcn_wmma_f32_16x16x32_bf16(false, fq.v, false, fb1.v, (short)0, z, false, false);
    }

    // Online softmax in C-layout registers; row reductions stay inside the
    // 16-lane half that owns the row (shfl_xor masks 1,2,4,8).
    int rbase = (lane & 16) ? 8 : 0;
    #pragma unroll
    for (int r = 0; r < 8; ++r) {
      float x0 = s0[r], x1 = s1[r];
      float mx = fmaxf(x0, x1);
      #pragma unroll
      for (int msk = 1; msk < 16; msk <<= 1)
        mx = fmaxf(mx, __shfl_xor(mx, msk, 32));
      float mnew  = fmaxf(m_st[r], mx);
      float alpha = __expf(m_st[r] - mnew);
      float p0 = __expf(x0 - mnew);
      float p1 = __expf(x1 - mnew);
      float sum = p0 + p1;
      #pragma unroll
      for (int msk = 1; msk < 16; msk <<= 1)
        sum += __shfl_xor(sum, msk, 32);
      l_st[r] = l_st[r] * alpha + sum;
      m_st[r] = mnew;
      o0[r] *= alpha;
      o1[r] *= alpha;
      sP[w][(r + rbase) * LPAD + col]      = f2bf(p0);
      sP[w][(r + rbase) * LPAD + 16 + col] = f2bf(p1);
    }
    asm volatile("s_wait_dscnt 0" ::: "memory");

    // O += P @ V (A = P from [qrow][key], B = V from [d][key] transposed)
    {
      FragBF fp, fv0, fv1;
      fp.q[0]  = *(const uiv4*)&sP[w][col * LPAD + kb];
      fp.q[1]  = *(const uiv4*)&sP[w][col * LPAD + 16 + kb];
      fv0.q[0] = *(const uiv4*)&sVt[w][col * LPAD + kb];
      fv0.q[1] = *(const uiv4*)&sVt[w][col * LPAD + 16 + kb];
      fv1.q[0] = *(const uiv4*)&sVt[w][(16 + col) * LPAD + kb];
      fv1.q[1] = *(const uiv4*)&sVt[w][(16 + col) * LPAD + 16 + kb];
      o0 = __builtin_amdgcn_wmma_f32_16x16x32_bf16(false, fp.v, false, fv0.v, (short)0, o0, false, false);
      o1 = __builtin_amdgcn_wmma_f32_16x16x32_bf16(false, fp.v, false, fv1.v, (short)0, o1, false, false);
    }
  }

  // Normalize and store (C layout)
  #pragma unroll
  for (int r = 0; r < 8; ++r) {
    int q = qt * 16 + r + ((lane & 16) ? 8 : 0);
    float inv = 1.f / l_st[r];
    O[base + (size_t)q * D_ + col]      = o0[r] * inv;
    O[base + (size_t)q * D_ + 16 + col] = o1[r] * inv;
  }
}

// ---------------------------------------------------------------------------
// out[row] = LN(res[row] + add[row] + bias?) * gamma + beta  (one block/row)
// ---------------------------------------------------------------------------
__global__ __launch_bounds__(256)
void add_ln_kernel(const float* __restrict__ res, const float* __restrict__ add,
                   const float* __restrict__ bias,
                   const float* __restrict__ gamma, const float* __restrict__ beta,
                   float* __restrict__ out) {
  __shared__ float red[8], red2[8];
  const int tid = threadIdx.x;
  const int lane = tid & 31, w = tid >> 5;
  const size_t rowoff = (size_t)blockIdx.x * D_;
  float v = res[rowoff + tid] + add[rowoff + tid] + (bias ? bias[tid] : 0.f);
  float s = v;
  #pragma unroll
  for (int msk = 1; msk < 32; msk <<= 1) s += __shfl_xor(s, msk, 32);
  if (lane == 0) red[w] = s;
  __syncthreads();
  float tot = 0.f;
  #pragma unroll
  for (int i = 0; i < 8; ++i) tot += red[i];
  float mu = tot * (1.f / (float)D_);
  float d = v - mu;
  float s2 = d * d;
  #pragma unroll
  for (int msk = 1; msk < 32; msk <<= 1) s2 += __shfl_xor(s2, msk, 32);
  if (lane == 0) red2[w] = s2;
  __syncthreads();
  float tot2 = 0.f;
  #pragma unroll
  for (int i = 0; i < 8; ++i) tot2 += red2[i];
  float var = tot2 * (1.f / (float)D_);
  out[rowoff + tid] = d * rsqrtf(var + EPS_) * gamma[tid] + beta[tid];
}

// ---------------------------------------------------------------------------
// Graph conv support kernels
// ---------------------------------------------------------------------------
__global__ void fill_zero(float* __restrict__ p, int n) {
  int i = blockIdx.x * blockDim.x + threadIdx.x;
  if (i < n) p[i] = 0.f;
}

__global__ void deg_kernel(const long long* __restrict__ eidx, float* __restrict__ deg) {
  int e = blockIdx.x * blockDim.x + threadIdx.x;
  if (e < E_) atomicAdd(&deg[eidx[e]], 1.f);
}

__global__ void dis_kernel(float* __restrict__ deg) {
  int i = blockIdx.x * blockDim.x + threadIdx.x;
  float dg = deg[i];
  deg[i] = (dg > 0.f) ? rsqrtf(dg) : 0.f;   // in-place: deg -> deg^-0.5
}

// agg[col] += dis[row]*dis[col] * xt[row] ; 64 threads per edge, 4 floats/thread
__global__ __launch_bounds__(256)
void scatter_kernel(const long long* __restrict__ eidx, const float* __restrict__ dis,
                    const float* __restrict__ xt, float* __restrict__ agg) {
  int e = blockIdx.x * 4 + (threadIdx.x >> 6);
  int t = threadIdx.x & 63;
  long long r = eidx[e];
  long long c = eidx[E_ + e];
  float nrm = dis[r] * dis[c];
  const float* src = xt + (size_t)r * D_;
  float* dst = agg + (size_t)c * D_;
  int d = t * 4;
  #pragma unroll
  for (int j = 0; j < 4; ++j)
    atomicAdd(&dst[d + j], nrm * src[d + j]);
}

// ---------------------------------------------------------------------------
extern "C" void kernel_launch(void* const* d_in, const int* in_sizes, int n_in,
                              void* d_out, int out_size, void* d_ws, size_t ws_size,
                              hipStream_t stream) {
  (void)in_sizes; (void)n_in; (void)out_size; (void)ws_size;
  const float*     x    = (const float*)d_in[0];
  const long long* ei   = (const long long*)d_in[1];
  const float* gc_w = (const float*)d_in[2];
  const float* gc_b = (const float*)d_in[3];
  const float* g_g  = (const float*)d_in[4];
  const float* g_b  = (const float*)d_in[5];
  const float* wq = (const float*)d_in[6];   const float* bq = (const float*)d_in[7];
  const float* wk = (const float*)d_in[8];   const float* bk = (const float*)d_in[9];
  const float* wv = (const float*)d_in[10];  const float* bv = (const float*)d_in[11];
  const float* wo = (const float*)d_in[12];  const float* bo = (const float*)d_in[13];
  const float* a_g = (const float*)d_in[14]; const float* a_b = (const float*)d_in[15];
  const float* w1 = (const float*)d_in[16];  const float* b1 = (const float*)d_in[17];
  const float* w2 = (const float*)d_in[18];  const float* b2 = (const float*)d_in[19];
  const float* m_g = (const float*)d_in[20]; const float* m_b = (const float*)d_in[21];
  float* out = (float*)d_out;

  // Workspace carve: 6 activation units of BNROWS*D floats + deg array.
  // p0: xt -> attnout -> (hbuf part), p1: agg -> ao -> (hbuf), p2: x1 -> (hbuf),
  // p3: q -> (hbuf), p4: k -> mlpout, p5: v -> x2.  hbuf = p0..p3 (contiguous 4U).
  float* ws = (float*)d_ws;
  const size_t U = (size_t)BNROWS * D_;
  float* p0 = ws + 0 * U;
  float* p1 = ws + 1 * U;
  float* p2 = ws + 2 * U;
  float* p3 = ws + 3 * U;
  float* p4 = ws + 4 * U;
  float* p5 = ws + 5 * U;
  float* deg = ws + 6 * U;

  dim3 blk(256);
  // 1) xt = x @ gc_w            -> p0
  gemm_bias_act<<<dim3(D_ / 64, BNROWS / 128), blk, 0, stream>>>(x, gc_w, nullptr, p0, BNROWS, D_, D_, 0);
  // 2) zero agg + deg; degrees; dis = deg^-0.5; scatter -> p1
  fill_zero<<<(int)((U + 255) / 256), blk, 0, stream>>>(p1, (int)U);
  fill_zero<<<(BNROWS + 255) / 256, blk, 0, stream>>>(deg, BNROWS);
  deg_kernel<<<E_ / 256, blk, 0, stream>>>(ei, deg);
  dis_kernel<<<BNROWS / 256, blk, 0, stream>>>(deg);
  scatter_kernel<<<E_ / 4, blk, 0, stream>>>(ei, deg, p0, p1);
  // 3) x1 = LN(x + agg + gc_b)  -> p2
  add_ln_kernel<<<BNROWS, blk, 0, stream>>>(x, p1, gc_b, g_g, g_b, p2);
  // 4) q,k,v projections        -> p3,p4,p5
  gemm_bias_act<<<dim3(D_ / 64, BNROWS / 128), blk, 0, stream>>>(p2, wq, bq, p3, BNROWS, D_, D_, 0);
  gemm_bias_act<<<dim3(D_ / 64, BNROWS / 128), blk, 0, stream>>>(p2, wk, bk, p4, BNROWS, D_, D_, 0);
  gemm_bias_act<<<dim3(D_ / 64, BNROWS / 128), blk, 0, stream>>>(p2, wv, bv, p5, BNROWS, D_, D_, 0);
  // 5) attention                -> p0
  attn_kernel<<<(B_ * H_ * (N_ / 16)) / 8, blk, 0, stream>>>(p3, p4, p5, p0);
  // 6) ao = attnout @ wo + bo   -> p1
  gemm_bias_act<<<dim3(D_ / 64, BNROWS / 128), blk, 0, stream>>>(p0, wo, bo, p1, BNROWS, D_, D_, 0);
  // 7) x2 = LN(x1 + ao)         -> p5
  add_ln_kernel<<<BNROWS, blk, 0, stream>>>(p2, p1, nullptr, a_g, a_b, p5);
  // 8) h = gelu(x2 @ w1 + b1)   -> hbuf (p0, 4 units)
  gemm_bias_act<<<dim3(MLPH_ / 64, BNROWS / 128), blk, 0, stream>>>(p5, w1, b1, p0, BNROWS, D_, MLPH_, 1);
  // 9) mlp = h @ w2 + b2        -> p4
  gemm_bias_act<<<dim3(D_ / 64, BNROWS / 128), blk, 0, stream>>>(p0, w2, b2, p4, BNROWS, MLPH_, D_, 0);
  // 10) out = LN(x2 + mlp)
  add_ln_kernel<<<BNROWS, blk, 0, stream>>>(p5, p4, nullptr, m_g, m_b, out);
}
